// MultiLSTM_41970420418187
// MI455X (gfx1250) — matrix operand order repaired and checked by reading, not compiled, not run on hardware
//
#include <hip/hip_runtime.h>
#include <hip/hip_bf16.h>

// ---------------------------------------------------------------------------
// MultiLSTM (3x LayerNorm-LSTM cells + linear head) for gfx1250 (MI455X).
// GEMMs run on V_WMMA_F32_16X16X32_BF16 (wave32), f32 accumulate.
// All streaming accesses forced to addrspace(1) => global_load/store_b128
// (FLAT ops would charge both LOADcnt and DScnt and use the LDS path).
// ---------------------------------------------------------------------------

#define B_DIM 8192
#define V_DIM 128
#define H_DIM 1024
#define G_DIM 4096   // 4*H
#define LN_EPS 1e-5f

#define GAS __attribute__((address_space(1)))

typedef __attribute__((ext_vector_type(16))) __bf16 v16bf;
typedef __attribute__((ext_vector_type(8)))  float  v8f;
typedef __attribute__((ext_vector_type(4)))  unsigned int u32x4;

union FragU { u32x4 u[2]; v16bf v; };

template <typename T>
__device__ __forceinline__ GAS T* gptr(T* p) {
  return (GAS T*)(unsigned long long)p;
}
template <typename T>
__device__ __forceinline__ GAS const T* gptr(const T* p) {
  return (GAS const T*)(unsigned long long)p;
}

__device__ __forceinline__ unsigned short f2bf(float f) {
  unsigned int u = __float_as_uint(f);
  unsigned int r = u + 0x7FFFu + ((u >> 16) & 1u);   // round-to-nearest-even
  return (unsigned short)(r >> 16);
}

__device__ __forceinline__ float sigmoidf(float x) {
  return 1.0f / (1.0f + __expf(-x));
}

// ---------------------------------------------------------------------------
// fp32 -> bf16 convert (grid-stride)
// ---------------------------------------------------------------------------
__global__ void f32_to_bf16_kernel(const float* __restrict__ in,
                                   unsigned short* __restrict__ out, int n) {
  GAS const float* gin = gptr(in);
  GAS unsigned short* gout = gptr(out);
  int i = blockIdx.x * blockDim.x + threadIdx.x;
  int stride = gridDim.x * blockDim.x;
  for (; i < n; i += stride) gout[i] = f2bf(gin[i]);
}

// ---------------------------------------------------------------------------
// C[M,N](f32) = A[M,K](bf16, row-major) x Bt[N,K](bf16, row-major)^T  (+bias)
// Block: 128 threads = 4 waves; block tile 64(M) x 128(N); wave tile 32x64.
// Fragment layouts follow CDNA5 ISA 7.12.2 (16-bit A 16x32, B 32x16, f32 C).
// Requires M%64==0, N%128==0, K%32==0 (true for all launches here).
// ---------------------------------------------------------------------------
__global__ __launch_bounds__(128) void gemm_nt_bf16_wmma(
    const unsigned short* __restrict__ A,
    const unsigned short* __restrict__ Bt,
    float* __restrict__ C,
    const float* __restrict__ bias,
    int M, int N, int K) {
  GAS const unsigned short* Ag = gptr(A);
  GAS const unsigned short* Bg = gptr(Bt);
  GAS float* Cg = gptr(C);

  const int lane = threadIdx.x & 31;
  const int wave = threadIdx.x >> 5;
  const int rowBase = blockIdx.y * 64 + (wave >> 1) * 32;
  const int colBase = blockIdx.x * 128 + (wave & 1) * 64;
  const int lr = lane & 15;   // row/col within 16x16 tile
  const int lh = lane >> 4;   // half-wave select

  v8f acc[2][4];
#pragma unroll
  for (int i = 0; i < 2; ++i)
#pragma unroll
    for (int j = 0; j < 4; ++j)
#pragma unroll
      for (int e = 0; e < 8; ++e) acc[i][j][e] = 0.0f;

  // A fragment: lane holds row (rowBase+i*16+lr), K = [kb..kb+7, 16+kb..16+kb+7]
  GAS const unsigned short* aPtr[2];
#pragma unroll
  for (int i = 0; i < 2; ++i)
    aPtr[i] = Ag + (size_t)(rowBase + i * 16 + lr) * K + lh * 8;
  // B fragment: lane holds col (colBase+j*16+lr), K = [kb..kb+7, 16+kb..16+kb+7]
  GAS const unsigned short* bPtr[4];
#pragma unroll
  for (int j = 0; j < 4; ++j)
    bPtr[j] = Bg + (size_t)(colBase + j * 16 + lr) * K + lh * 8;

  for (int kk = 0; kk < K; kk += 32) {
    if (kk + 32 < K) {
      __builtin_prefetch(aPtr[0] + 32, 0, 3);  // prefetch next K slab (near scope)
      __builtin_prefetch(bPtr[0] + 32, 0, 3);
    }
    FragU af[2], bf[4];
#pragma unroll
    for (int i = 0; i < 2; ++i) {
      af[i].u[0] = *(GAS const u32x4*)(aPtr[i]);        // K kb..kb+7
      af[i].u[1] = *(GAS const u32x4*)(aPtr[i] + 16);   // K 16+kb..16+kb+7
      aPtr[i] += 32;
    }
#pragma unroll
    for (int j = 0; j < 4; ++j) {
      bf[j].u[0] = *(GAS const u32x4*)(bPtr[j]);        // K kb..kb+7
      bf[j].u[1] = *(GAS const u32x4*)(bPtr[j] + 16);   // K 16+kb..16+kb+7
      bPtr[j] += 32;
    }
#pragma unroll
    for (int i = 0; i < 2; ++i)
#pragma unroll
      for (int j = 0; j < 4; ++j)
        acc[i][j] = __builtin_amdgcn_wmma_f32_16x16x32_bf16(
            false, af[i].v, false, bf[j].v, (short)0, acc[i][j], false, false);
  }

  // C/D layout: element e -> row (i*16 + lh*8 + e), col (j*16 + lr)
#pragma unroll
  for (int i = 0; i < 2; ++i) {
#pragma unroll
    for (int j = 0; j < 4; ++j) {
      const int col = colBase + j * 16 + lr;
      const float bv = bias ? bias[col] : 0.0f;
#pragma unroll
      for (int e = 0; e < 8; ++e) {
        const int row = rowBase + i * 16 + lh * 8 + e;
        Cg[(size_t)row * N + col] = acc[i][j][e] + bv;
      }
    }
  }
}

// ---------------------------------------------------------------------------
// Per-row fused LSTM cell: LN(Gi)+LN(Gh) -> gates -> LN(c) -> hy/cy.
// One block (256 threads, 8 wave32) per batch row.
// ---------------------------------------------------------------------------
__device__ __forceinline__ float blockReduceSum(float v, volatile float* sm,
                                                int tid) {
#pragma unroll
  for (int o = 16; o > 0; o >>= 1) v += __shfl_xor(v, o, 32);
  const int wid = tid >> 5;
  if ((tid & 31) == 0) sm[wid] = v;
  __syncthreads();
  float t = 0.0f;
#pragma unroll
  for (int i = 0; i < 8; ++i) t += sm[i];
  __syncthreads();
  return t;
}

__global__ __launch_bounds__(256) void lstm_cell_pointwise(
    const float* __restrict__ Gi, const float* __restrict__ Gh,
    const float* __restrict__ cx,
    const float* __restrict__ lni_w, const float* __restrict__ lni_b,
    const float* __restrict__ lnh_w, const float* __restrict__ lnh_b,
    const float* __restrict__ lnc_w, const float* __restrict__ lnc_b,
    float* __restrict__ hy, float* __restrict__ cy,
    unsigned short* __restrict__ hy_bf) {
  const int b = blockIdx.x;
  const int tid = threadIdx.x;
  GAS const float* gi_row = gptr(Gi) + (size_t)b * G_DIM;
  GAS const float* gh_row = gptr(Gh) + (size_t)b * G_DIM;
  GAS const float* cx_row = gptr(cx) + (size_t)b * H_DIM;
  GAS const float* wi = gptr(lni_w);
  GAS const float* bi = gptr(lni_b);
  GAS const float* wh = gptr(lnh_w);
  GAS const float* bh = gptr(lnh_b);
  GAS const float* wc = gptr(lnc_w);
  GAS const float* bc = gptr(lnc_b);
  GAS float* hy_g = gptr(hy);
  GAS float* cy_g = gptr(cy);
  GAS unsigned short* hybf_g = gptr(hy_bf);

  __shared__ float sm0[8], sm1[8], sm2[8], sm3[8], sm4[8], sm5[8];

  float gi[16], gh[16];
  float si = 0.f, ssi = 0.f, sh = 0.f, ssh = 0.f;
#pragma unroll
  for (int t = 0; t < 16; ++t) {
    const int j = tid + 256 * t;
    const float a = gi_row[j];
    const float h = gh_row[j];
    gi[t] = a; gh[t] = h;
    si += a; ssi += a * a; sh += h; ssh += h * h;
  }
  si  = blockReduceSum(si,  sm0, tid);
  ssi = blockReduceSum(ssi, sm1, tid);
  sh  = blockReduceSum(sh,  sm2, tid);
  ssh = blockReduceSum(ssh, sm3, tid);

  const float inv4h = 1.0f / (float)G_DIM;
  const float mi = si * inv4h, mh = sh * inv4h;
  const float ri = rsqrtf(ssi * inv4h - mi * mi + LN_EPS);
  const float rh = rsqrtf(ssh * inv4h - mh * mh + LN_EPS);

#pragma unroll
  for (int t = 0; t < 16; ++t) {
    const int j = tid + 256 * t;
    gi[t] = (gi[t] - mi) * ri * wi[j] + bi[j] +
            (gh[t] - mh) * rh * wh[j] + bh[j];
  }

  // split: t=0..3 -> i gate, 4..7 -> f, 8..11 -> g, 12..15 -> o
  float cpre[4], og[4];
  float sc = 0.f, ssc = 0.f;
#pragma unroll
  for (int u = 0; u < 4; ++u) {
    const int idx = tid + 256 * u;
    const float iv = sigmoidf(gi[u]);
    const float fv = sigmoidf(gi[4 + u]);
    const float gv = tanhf(gi[8 + u]);
    og[u] = gi[12 + u];
    const float c = fv * cx_row[idx] + iv * gv;
    cpre[u] = c; sc += c; ssc += c * c;
  }
  sc  = blockReduceSum(sc,  sm4, tid);
  ssc = blockReduceSum(ssc, sm5, tid);
  const float invh = 1.0f / (float)H_DIM;
  const float mc = sc * invh;
  const float rc = rsqrtf(ssc * invh - mc * mc + LN_EPS);

#pragma unroll
  for (int u = 0; u < 4; ++u) {
    const int idx = tid + 256 * u;
    const float cyv = (cpre[u] - mc) * rc * wc[idx] + bc[idx];
    const float hyv = sigmoidf(og[u]) * tanhf(cyv);
    const size_t o = (size_t)b * H_DIM + idx;
    cy_g[o] = cyv;
    hy_g[o] = hyv;
    hybf_g[o] = f2bf(hyv);
  }
}

// ---------------------------------------------------------------------------
// Launch
// ---------------------------------------------------------------------------
extern "C" void kernel_launch(void* const* d_in, const int* in_sizes, int n_in,
                              void* d_out, int out_size, void* d_ws,
                              size_t ws_size, hipStream_t stream) {
  (void)in_sizes; (void)n_in; (void)out_size; (void)ws_size;

  const float* x     = (const float*)d_in[0];
  const float* h0    = (const float*)d_in[1];
  const float* c0    = (const float*)d_in[2];
  const float* h1    = (const float*)d_in[3];
  const float* c1    = (const float*)d_in[4];
  const float* h2    = (const float*)d_in[5];
  const float* c2    = (const float*)d_in[6];
  const float* w_ih1 = (const float*)d_in[7];
  const float* w_hh1 = (const float*)d_in[8];
  const float* w_ih2 = (const float*)d_in[9];
  const float* w_hh2 = (const float*)d_in[10];
  const float* w_ih3 = (const float*)d_in[11];
  const float* w_hh3 = (const float*)d_in[12];
  // ln params: base 13 + (l-1)*6: lni_w, lni_b, lnh_w, lnh_b, lnc_w, lnc_b
  const float* ln[3][6];
  for (int l = 0; l < 3; ++l)
    for (int q = 0; q < 6; ++q) ln[l][q] = (const float*)d_in[13 + l * 6 + q];
  const float* lin_w = (const float*)d_in[31];
  const float* lin_b = (const float*)d_in[32];

  float* out    = (float*)d_out;
  float* logits = out;
  float* hy1    = logits + (size_t)B_DIM * V_DIM;
  float* cy1    = hy1 + (size_t)B_DIM * H_DIM;
  float* hy2    = cy1 + (size_t)B_DIM * H_DIM;
  float* cy2    = hy2 + (size_t)B_DIM * H_DIM;
  float* hy3    = cy2 + (size_t)B_DIM * H_DIM;
  float* cy3    = hy3 + (size_t)B_DIM * H_DIM;

  // workspace carve-out (256B aligned)
  char* wsp = (char*)d_ws;
  auto alloc = [&](size_t bytes) -> void* {
    void* p = (void*)wsp;
    wsp += (bytes + 255) & ~(size_t)255;
    return p;
  };
  unsigned short* xbf    = (unsigned short*)alloc((size_t)B_DIM * V_DIM * 2);
  unsigned short* h0bf   = (unsigned short*)alloc((size_t)B_DIM * H_DIM * 2);
  unsigned short* h1bf   = (unsigned short*)alloc((size_t)B_DIM * H_DIM * 2);
  unsigned short* h2bf   = (unsigned short*)alloc((size_t)B_DIM * H_DIM * 2);
  unsigned short* hy1bf  = (unsigned short*)alloc((size_t)B_DIM * H_DIM * 2);
  unsigned short* hy2bf  = (unsigned short*)alloc((size_t)B_DIM * H_DIM * 2);
  unsigned short* hy3bf  = (unsigned short*)alloc((size_t)B_DIM * H_DIM * 2);
  unsigned short* wih1bf = (unsigned short*)alloc((size_t)G_DIM * V_DIM * 2);
  unsigned short* whh1bf = (unsigned short*)alloc((size_t)G_DIM * H_DIM * 2);
  unsigned short* wih2bf = (unsigned short*)alloc((size_t)G_DIM * H_DIM * 2);
  unsigned short* whh2bf = (unsigned short*)alloc((size_t)G_DIM * H_DIM * 2);
  unsigned short* wih3bf = (unsigned short*)alloc((size_t)G_DIM * H_DIM * 2);
  unsigned short* whh3bf = (unsigned short*)alloc((size_t)G_DIM * H_DIM * 2);
  unsigned short* linwbf = (unsigned short*)alloc((size_t)V_DIM * H_DIM * 2);
  float* Gi = (float*)alloc((size_t)B_DIM * G_DIM * 4);
  float* Gh = (float*)alloc((size_t)B_DIM * G_DIM * 4);

  auto cvt = [&](const float* src, unsigned short* dst, int n) {
    int blocks = (n + 256 * 16 - 1) / (256 * 16);
    if (blocks > 2048) blocks = 2048;
    f32_to_bf16_kernel<<<blocks, 256, 0, stream>>>(src, dst, n);
  };
  cvt(x, xbf, B_DIM * V_DIM);
  cvt(h0, h0bf, B_DIM * H_DIM);
  cvt(h1, h1bf, B_DIM * H_DIM);
  cvt(h2, h2bf, B_DIM * H_DIM);
  cvt(w_ih1, wih1bf, G_DIM * V_DIM);
  cvt(w_hh1, whh1bf, G_DIM * H_DIM);
  cvt(w_ih2, wih2bf, G_DIM * H_DIM);
  cvt(w_hh2, whh2bf, G_DIM * H_DIM);
  cvt(w_ih3, wih3bf, G_DIM * H_DIM);
  cvt(w_hh3, whh3bf, G_DIM * H_DIM);
  cvt(lin_w, linwbf, V_DIM * H_DIM);

  auto gemm = [&](const unsigned short* A, const unsigned short* Bt, float* C,
                  const float* bias, int M, int N, int K) {
    dim3 grid(N / 128, M / 64);
    gemm_nt_bf16_wmma<<<grid, dim3(128), 0, stream>>>(A, Bt, C, bias, M, N, K);
  };
  auto cell = [&](const float* cx, int l, float* hyo, float* cyo,
                  unsigned short* hybf) {
    lstm_cell_pointwise<<<B_DIM, 256, 0, stream>>>(
        Gi, Gh, cx, ln[l][0], ln[l][1], ln[l][2], ln[l][3], ln[l][4], ln[l][5],
        hyo, cyo, hybf);
  };

  // layer 1
  gemm(xbf,  wih1bf, Gi, nullptr, B_DIM, G_DIM, V_DIM);
  gemm(h0bf, whh1bf, Gh, nullptr, B_DIM, G_DIM, H_DIM);
  cell(c0, 0, hy1, cy1, hy1bf);
  // layer 2
  gemm(hy1bf, wih2bf, Gi, nullptr, B_DIM, G_DIM, H_DIM);
  gemm(h1bf,  whh2bf, Gh, nullptr, B_DIM, G_DIM, H_DIM);
  cell(c1, 1, hy2, cy2, hy2bf);
  // layer 3
  gemm(hy2bf, wih3bf, Gi, nullptr, B_DIM, G_DIM, H_DIM);
  gemm(h2bf,  whh3bf, Gh, nullptr, B_DIM, G_DIM, H_DIM);
  cell(c2, 2, hy3, cy3, hy3bf);
  // head
  gemm(hy3bf, linwbf, logits, lin_b, B_DIM, V_DIM, H_DIM);
}
